// GNNLayer_65627100283535
// MI455X (gfx1250) — matrix operand order repaired and checked by reading, not compiled, not run on hardware
//
#include <hip/hip_runtime.h>
#include <hip/hip_bf16.h>
#include <math.h>

// ---------------------------------------------------------------------------
// GNN edge-attention layer for gfx1250 (MI455X), wave32 + WMMA + async-to-LDS.
//   prep: Wp,Wm,W1,W2 -> f16 copies in workspace (once per launch)
//   q  = x[src] @ Wp^T + bp          (WMMA f16->f32, K=128; B from LDS via
//                                     global_load_async_to_lds_b128)
//   kv = x[dst] @ Wp^T + bp          (WMMA)
//   attn: 16x16 scores, K=H=8        (fp32 VALU + softmax)
//   msg = attn_out @ Wm^T + bm       (WMMA; B = f16 global, L2 resident)
//   agg = segment_sum(msg, col)      (global_atomic_add_f32 into L2-resident ws)
//   h   = relu([x|agg] @ W1^T + b1)  (WMMA, K=256; B = f16 global)
//   out = h @ W2^T + b2              (WMMA, K=128; B from LDS via async)
// ---------------------------------------------------------------------------

#define D_  128
#define H_  8
#define WAVES_PB 4   // 128 threads / block

typedef __attribute__((ext_vector_type(16))) _Float16 v16h;
typedef __attribute__((ext_vector_type(8)))  float    v8f;

// K index of VGPR pair v for a 16x32 f16 A fragment (contiguous order, matches B):
// lanes 0-15: V0..7 -> K 0..15 ; lanes 16-31: +16
__device__ __forceinline__ int a_k_of(int v, int half) {
  return 2 * v + (half ? 16 : 0);
}

// Build A fragment (this lane's matrix row) from an f32 row pointer.
__device__ __forceinline__ v16h load_a_f32row(const float* __restrict__ rowPtr,
                                              int kbase, int half) {
  v16h a;
#pragma unroll
  for (int v = 0; v < 8; ++v) {
    int k = kbase + a_k_of(v, half);
    float2 f = *reinterpret_cast<const float2*>(rowPtr + k);
    a[2 * v]     = (_Float16)f.x;
    a[2 * v + 1] = (_Float16)f.y;
  }
  return a;
}

// Build A fragment from an f16 row (LDS).
__device__ __forceinline__ v16h load_a_h16row(const _Float16* __restrict__ rowPtr,
                                              int kbase, int half) {
  v16h a;
#pragma unroll
  for (int v = 0; v < 8; ++v) {
    int k = kbase + a_k_of(v, half);
    a[2 * v]     = rowPtr[k];
    a[2 * v + 1] = rowPtr[k + 1];
  }
  return a;
}

// B fragment for B[k][n] = W16[ncol][kbase+k]; the lane's 16 values are
// contiguous along k, 32B-aligned -> single v16h load (2x b128).
__device__ __forceinline__ v16h load_b_h16(const _Float16* W16, int ld,
                                           int ncol, int kbase, int half) {
  const _Float16* p = W16 + (size_t)ncol * ld + kbase + (half ? 16 : 0);
  return *reinterpret_cast<const v16h*>(p);
}

__device__ __forceinline__ v8f wmma16(v16h a, v16h b, v8f c) {
  return __builtin_amdgcn_wmma_f32_16x16x32_f16(false, a, false, b,
                                                (short)0, c, false, false);
}

// gfx1250 async global->LDS copy (ASYNCcnt), ISA 10.x / 15.18.3.
__device__ __forceinline__ void async_ld_16B(unsigned int lds_off, const void* gptr) {
  asm volatile("global_load_async_to_lds_b128 %0, %1, off"
               :: "v"(lds_off), "v"(gptr) : "memory");
}
__device__ __forceinline__ void wait_async0() {
  asm volatile("s_wait_asynccnt 0x0" ::: "memory");
}

// ---------------------------------------------------------------------------
__global__ void gnn_zero_kernel(float* __restrict__ p, long long n) {
  long long i = (long long)blockIdx.x * blockDim.x + threadIdx.x;
  if (i < n) p[i] = 0.0f;
}

// Convert the four weight matrices to f16 once per launch.
__global__ void gnn_prep_kernel(const float* __restrict__ Wp, const float* __restrict__ Wm,
                                const float* __restrict__ W1, const float* __restrict__ W2,
                                _Float16* __restrict__ wpH, _Float16* __restrict__ wmH,
                                _Float16* __restrict__ w1H, _Float16* __restrict__ w2H) {
  int i = blockIdx.x * blockDim.x + threadIdx.x;
  if (i < D_ * D_)     wpH[i] = (_Float16)Wp[i];
  if (i < D_ * D_)     wmH[i] = (_Float16)Wm[i];
  if (i < D_ * 2 * D_) w1H[i] = (_Float16)W1[i];
  if (i < D_ * D_)     w2H[i] = (_Float16)W2[i];
}

// ---------------------------------------------------------------------------
// One wave processes 16 edges.
__global__ __launch_bounds__(WAVES_PB * 32)
void gnn_edge_kernel(const float* __restrict__ x,
                     const int* __restrict__ edges,   // (E,2) interleaved
                     const _Float16* __restrict__ wpH, const float* __restrict__ bp,
                     const _Float16* __restrict__ wmH, const float* __restrict__ bm,
                     float* __restrict__ agg, int E, int N) {
  __shared__ alignas(32) _Float16 s_wp[D_ * D_];            // 32 KB (async-filled)
  __shared__ alignas(32) _Float16 s_q [WAVES_PB][16 * D_];  // 16 KB (out aliases q)
  __shared__ alignas(32) _Float16 s_kv[WAVES_PB][16 * D_];  // 16 KB

  const int wave = threadIdx.x >> 5;
  const int lane = threadIdx.x & 31;
  const int half = lane >> 4;
  const int r    = lane & 15;

  // ---- stage Wp(f16) into LDS with async copies; all threads participate --
  {
    const unsigned int base = (unsigned int)(uintptr_t)(&s_wp[0]);
    const char* g = (const char*)wpH;
#pragma unroll
    for (int i = threadIdx.x; i < (D_ * D_ * 2) / 16; i += WAVES_PB * 32)
      async_ld_16B(base + (unsigned int)i * 16u, g + (size_t)i * 16);
    wait_async0();
  }
  __syncthreads();

  long long tile = (long long)blockIdx.x * WAVES_PB + wave;
  const int eBase = (int)(tile * 16);
  if (eBase >= E) return;                       // wave-uniform exit (after barrier)

  _Float16* qh  = s_q[wave];
  _Float16* kvh = s_kv[wave];
  _Float16* oh  = qh;   // attention output overwrites q rows (same-wave, in-order DS)

  __builtin_prefetch(edges + 2 * (size_t)(eBase + 16), 0, 0);

  // ---- gather A fragments for src/dst rows (row = lane%16) ----------------
  int eRow = eBase + r;  if (eRow >= E) eRow = E - 1;
  const int srcIdx = edges[2 * (size_t)eRow];
  const int dstIdx = edges[2 * (size_t)eRow + 1];
  const float* srcRow = x + (size_t)srcIdx * D_;
  const float* dstRow = x + (size_t)dstIdx * D_;

  v16h aS[4], aD[4];
#pragma unroll
  for (int ks = 0; ks < 4; ++ks) {
    aS[ks] = load_a_f32row(srcRow, ks * 32, half);
    aD[ks] = load_a_f32row(dstRow, ks * 32, half);
  }

  // ---- q / kv projections (D x D GEMM, K = 128), B from LDS ---------------
#pragma unroll
  for (int nt = 0; nt < 8; ++nt) {
    const float bpv = bp[nt * 16 + r];
    v8f cq, ck;
#pragma unroll
    for (int i = 0; i < 8; ++i) { cq[i] = bpv; ck[i] = bpv; }
#pragma unroll
    for (int ks = 0; ks < 4; ++ks) {
      v16h b = load_b_h16(s_wp, D_, nt * 16 + r, ks * 32, half);
      cq = wmma16(aS[ks], b, cq);
      ck = wmma16(aD[ks], b, ck);
    }
#pragma unroll
    for (int i = 0; i < 8; ++i) {
      int m = i + 8 * half;                     // C-layout: M = vgpr + 8*half
      qh [m * D_ + nt * 16 + r] = (_Float16)cq[i];
      kvh[m * D_ + nt * 16 + r] = (_Float16)ck[i];
    }
  }

  // ---- per-edge attention: scores(16x16, K=H=8) -> softmax -> out ---------
  const float inv_sqrtH = 0.35355339059327373f; // 1/sqrt(8)
#pragma unroll 2
  for (int pass = 0; pass < 8; ++pass) {
    const int el = 2 * pass + half;
    const _Float16* qrow = qh + el * D_;
    const _Float16* krow = kvh + el * D_;

    float qv[8];
#pragma unroll
    for (int h = 0; h < H_; ++h) qv[h] = (float)qrow[h * 16 + r];

    float s[16];
    float mx = -1e30f;
#pragma unroll
    for (int m = 0; m < 16; ++m) {
      float acc = 0.0f;
#pragma unroll
      for (int h = 0; h < H_; ++h) acc += qv[h] * (float)krow[h * 16 + m];
      s[m] = acc * inv_sqrtH;
      mx = fmaxf(mx, s[m]);
    }
    float sum = 0.0f;
#pragma unroll
    for (int m = 0; m < 16; ++m) { s[m] = __expf(s[m] - mx); sum += s[m]; }
    const float rs = 1.0f / sum;

#pragma unroll
    for (int h = 0; h < H_; ++h) {
      float acc = 0.0f;
#pragma unroll
      for (int m = 0; m < 16; ++m) acc += s[m] * (float)krow[h * 16 + m];
      oh[el * D_ + h * 16 + r] = (_Float16)(acc * rs);   // overwrites q row `el`
    }
  }

  // ---- msg = out @ Wm^T + bm, then atomic segment-sum into agg ------------
  v16h aO[4];
#pragma unroll
  for (int ks = 0; ks < 4; ++ks)
    aO[ks] = load_a_h16row(oh + r * D_, ks * 32, half);

  int colIdx[8];
#pragma unroll
  for (int v = 0; v < 8; ++v) {
    int e = eBase + v + 8 * half;  if (e >= E) e = E - 1;
    colIdx[v] = edges[2 * (size_t)e + 1];
  }

#pragma unroll
  for (int nt = 0; nt < 8; ++nt) {
    const float bmv = bm[nt * 16 + r];
    v8f c;
#pragma unroll
    for (int i = 0; i < 8; ++i) c[i] = bmv;
#pragma unroll
    for (int ks = 0; ks < 4; ++ks) {
      v16h b = load_b_h16(wmH, D_, nt * 16 + r, ks * 32, half);
      c = wmma16(aO[ks], b, c);
    }
#pragma unroll
    for (int v = 0; v < 8; ++v)
      atomicAdd(&agg[(size_t)colIdx[v] * D_ + nt * 16 + r], c[v]);
  }
}

// ---------------------------------------------------------------------------
// One wave processes 16 nodes: h = relu([x|agg]@W1^T+b1); out = h@W2^T+b2.
__global__ __launch_bounds__(WAVES_PB * 32)
void gnn_node_kernel(const float* __restrict__ x, const float* __restrict__ agg,
                     const _Float16* __restrict__ w1H, const float* __restrict__ b1,
                     const _Float16* __restrict__ w2H, const float* __restrict__ b2,
                     float* __restrict__ out, int N) {
  __shared__ alignas(32) _Float16 s_w2[D_ * D_];            // 32 KB (async-filled)
  __shared__ alignas(32) _Float16 s_h1[WAVES_PB][16 * D_];  // 16 KB

  const int wave = threadIdx.x >> 5;
  const int lane = threadIdx.x & 31;
  const int half = lane >> 4;
  const int r    = lane & 15;

  // ---- stage W2(f16) into LDS with async copies ---------------------------
  {
    const unsigned int base = (unsigned int)(uintptr_t)(&s_w2[0]);
    const char* g = (const char*)w2H;
#pragma unroll
    for (int i = threadIdx.x; i < (D_ * D_ * 2) / 16; i += WAVES_PB * 32)
      async_ld_16B(base + (unsigned int)i * 16u, g + (size_t)i * 16);
    wait_async0();
  }
  __syncthreads();

  long long tile = (long long)blockIdx.x * WAVES_PB + wave;
  const int nBase = (int)(tile * 16);
  if (nBase >= N) return;                       // wave-uniform exit (after barrier)

  _Float16* h1 = s_h1[wave];

  int node = nBase + r;  if (node >= N) node = N - 1;
  const float* xrow = x   + (size_t)node * D_;
  const float* arow = agg + (size_t)node * D_;

  // A fragments over the concatenated 256-deep K dimension.
  v16h a1[8];
#pragma unroll
  for (int ks = 0; ks < 8; ++ks) {
    const int kb = ks * 32;
    const float* base = (kb < D_) ? (xrow + kb) : (arow + (kb - D_));
    a1[ks] = load_a_f32row(base, 0, half);
  }

  // ---- GEMM1: (16 x 256) @ (256 x 128) + ReLU, B = f16 global -------------
#pragma unroll
  for (int nt = 0; nt < 8; ++nt) {
    const float b1v = b1[nt * 16 + r];
    v8f c;
#pragma unroll
    for (int i = 0; i < 8; ++i) c[i] = b1v;
#pragma unroll
    for (int ks = 0; ks < 8; ++ks) {
      v16h b = load_b_h16(w1H, 2 * D_, nt * 16 + r, ks * 32, half);
      c = wmma16(a1[ks], b, c);
    }
#pragma unroll
    for (int i = 0; i < 8; ++i) {
      int m = i + 8 * half;
      h1[m * D_ + nt * 16 + r] = (_Float16)fmaxf(c[i], 0.0f);
    }
  }

  // ---- GEMM2: (16 x 128) @ (128 x 128) -> out, B from LDS -----------------
  v16h a2[4];
#pragma unroll
  for (int ks = 0; ks < 4; ++ks)
    a2[ks] = load_a_h16row(h1 + r * D_, ks * 32, half);

#pragma unroll
  for (int nt = 0; nt < 8; ++nt) {
    const float b2v = b2[nt * 16 + r];
    v8f c;
#pragma unroll
    for (int i = 0; i < 8; ++i) c[i] = b2v;
#pragma unroll
    for (int ks = 0; ks < 4; ++ks) {
      v16h b = load_b_h16(s_w2, D_, nt * 16 + r, ks * 32, half);
      c = wmma16(a2[ks], b, c);
    }
#pragma unroll
    for (int i = 0; i < 8; ++i) {
      int m = nBase + i + 8 * half;
      if (m < N) out[(size_t)m * D_ + nt * 16 + r] = c[i];
    }
  }
}

// ---------------------------------------------------------------------------
extern "C" void kernel_launch(void* const* d_in, const int* in_sizes, int n_in,
                              void* d_out, int out_size, void* d_ws, size_t ws_size,
                              hipStream_t stream) {
  const float* x     = (const float*)d_in[0];
  const int*   edges = (const int*)  d_in[1];
  const float* Wp    = (const float*)d_in[2];
  const float* bp    = (const float*)d_in[3];
  const float* Wm    = (const float*)d_in[4];
  const float* bm    = (const float*)d_in[5];
  const float* W1    = (const float*)d_in[6];
  const float* b1    = (const float*)d_in[7];
  const float* W2    = (const float*)d_in[8];
  const float* b2    = (const float*)d_in[9];
  float* out = (float*)d_out;

  const int N = in_sizes[0] / D_;
  const int E = in_sizes[1] / 2;

  // Workspace layout: [agg: N*D f32][wpH][wmH][w1H][w2H] (all 32B aligned).
  float* agg = (float*)d_ws;
  _Float16* wpH = (_Float16*)((char*)d_ws + (size_t)N * D_ * sizeof(float));
  _Float16* wmH = wpH + D_ * D_;
  _Float16* w1H = wmH + D_ * D_;
  _Float16* w2H = w1H + D_ * 2 * D_;

  // 1) zero the aggregation buffer + convert weights to f16 (independent)
  const long long nAgg = (long long)N * D_;
  gnn_zero_kernel<<<(int)((nAgg + 255) / 256), 256, 0, stream>>>(agg, nAgg);
  gnn_prep_kernel<<<(D_ * 2 * D_ + 255) / 256, 256, 0, stream>>>(
      Wp, Wm, W1, W2, wpH, wmH, w1H, w2H);

  // 2) edge attention + message + atomic segment-sum
  const int eTiles = (E + 15) / 16;
  gnn_edge_kernel<<<(eTiles + WAVES_PB - 1) / WAVES_PB, WAVES_PB * 32, 0, stream>>>(
      x, edges, wpH, bp, wmH, bm, agg, E, N);

  // 3) node MLP
  const int nTiles = (N + 15) / 16;
  gnn_node_kernel<<<(nTiles + WAVES_PB - 1) / WAVES_PB, WAVES_PB * 32, 0, stream>>>(
      x, agg, w1H, b1, w2H, b2, out, N);
}
